// Transformer_45629732552754
// MI455X (gfx1250) — matrix operand order repaired and checked by reading, not compiled, not run on hardware
//
#include <hip/hip_runtime.h>
#include <hip/hip_bf16.h>
#include <math.h>
#include <stdint.h>

// ---------------------------------------------------------------------------
// MI455X (gfx1250) transformer forward.
//  - All GEMMs on v_wmma_f32_16x16x32_bf16 (fp32 accumulate)
//  - Double-buffered LDS tiles, BK=64, one barrier per 16 WMMAs
//  - bf16 tile staging via GLOBAL_LOAD_ASYNC_TO_LDS_B128 (ASYNCcnt DMA)
//  - fp32 weights converted to bf16 during VGPR staging (overlapped)
// ---------------------------------------------------------------------------

typedef __bf16 bf16_t;
typedef bf16_t v16bf __attribute__((ext_vector_type(16)));
typedef bf16_t v8bf  __attribute__((ext_vector_type(8)));
typedef float  v8f   __attribute__((ext_vector_type(8)));

#define S_LEN 4096
#define D_DIM 1024
#define I_DIM 4096
#define L_NUM 4
#define C_DIM 1024

// ---- CDNA5 async global->LDS copy (16B per lane), tracked by ASYNCcnt ------
__device__ inline void async_b128(unsigned lds_addr, const void* gptr) {
  asm volatile("global_load_async_to_lds_b128 %0, %1, off"
               :: "v"(lds_addr), "v"((uint64_t)(size_t)gptr)
               : "memory");
}
__device__ inline void wait_async0() {
  asm volatile("s_wait_asynccnt 0x0" ::: "memory");
}

// ---------------------------- embedding ------------------------------------
__global__ __launch_bounds__(256) void embed_kernel(
    const int* __restrict__ idx, const float* __restrict__ emb,
    const float* __restrict__ pos, float* __restrict__ xF,
    bf16_t* __restrict__ xB)
{
  const int s = blockIdx.x;
  const int v = idx[s];
  for (int d = threadIdx.x; d < D_DIM; d += 256) {
    float val = emb[(size_t)v * D_DIM + d] + pos[(size_t)s * D_DIM + d];
    xF[(size_t)s * D_DIM + d] = val;
    xB[(size_t)s * D_DIM + d] = (bf16_t)val;
  }
}

__device__ inline v8bf cvt8_f32(const float* p) {
  float4 a = *(const float4*)p;
  float4 b = *(const float4*)(p + 4);
  v8bf r;
  r[0] = (bf16_t)a.x; r[1] = (bf16_t)a.y; r[2] = (bf16_t)a.z; r[3] = (bf16_t)a.w;
  r[4] = (bf16_t)b.x; r[5] = (bf16_t)b.y; r[6] = (bf16_t)b.z; r[7] = (bf16_t)b.w;
  return r;
}

// ---------------------------- WMMA GEMM ------------------------------------
// C[M,N] = epilogue( A[M,K](bf16) @ B[K,N] * scale + bias + resid )
// B fp32 or bf16; TRANS_B => B stored [N,K] row-major (q @ k^T).
// Block: 256 threads = 8 waves; tile 128x128, BK=64, double-buffered LDS.
// Wave grid 4x2; each wave owns 32x64 = 2x4 accum tiles of 16x16.
template<bool B_BF16, bool TRANS_B, bool RELU>
__global__ __launch_bounds__(256) void gemm_bf16_wmma(
    const bf16_t* __restrict__ A, const void* __restrict__ Bvp,
    const float* __restrict__ bias, const float* __restrict__ resid,
    float scale, float* __restrict__ outF, bf16_t* __restrict__ outB,
    int N, int K, int lda, int ldb)
{
  // 72-element row stride (144B): 16B-aligned, bank-conflict-friendly padding
  __shared__ bf16_t As[2][128][72];
  __shared__ bf16_t Bs[2][128][72];

  const bf16_t* Bb = (const bf16_t*)Bvp;
  const float*  Bf = (const float*)Bvp;

  const int tid   = threadIdx.x;
  const int lane  = tid & 31;
  const int wave  = tid >> 5;
  const int waveM = wave >> 1;     // 0..3
  const int waveN = wave & 1;      // 0..1
  const int half  = lane >> 4;     // 0/1
  const int lrow  = lane & 15;
  const int k0    = half * 8;      // A-frag K split per ISA 16-bit layout

  const int bm = blockIdx.y * 128;
  const int bn = blockIdx.x * 128;

  v8f acc[2][4];
  #pragma unroll
  for (int mi = 0; mi < 2; mi++)
    #pragma unroll
    for (int ni = 0; ni < 4; ni++)
      #pragma unroll
      for (int e = 0; e < 8; e++) acc[mi][ni][e] = 0.f;

  // ---- tile staging (issues async DMA / ds stores into buffer `buf`) ----
  auto stageA = [&](int buf, int kk) {
    const int r = tid >> 1, h = tid & 1;           // 64B contiguous per thread
    const bf16_t* src = A + (size_t)(bm + r) * lda + kk + h * 32;
    unsigned ld = (unsigned)(size_t)&As[buf][r][h * 32];
    async_b128(ld,      src);
    async_b128(ld + 16, src + 8);
    async_b128(ld + 32, src + 16);
    async_b128(ld + 48, src + 24);
  };
  auto stageB = [&](int buf, int kk) {
    if constexpr (TRANS_B) {
      const int r = tid >> 1, h = tid & 1;         // r = n; contiguous in k
      if constexpr (B_BF16) {
        const bf16_t* src = Bb + (size_t)(bn + r) * ldb + kk + h * 32;
        unsigned ld = (unsigned)(size_t)&Bs[buf][r][h * 32];
        async_b128(ld,      src);
        async_b128(ld + 16, src + 8);
        async_b128(ld + 32, src + 16);
        async_b128(ld + 48, src + 24);
      } else {
        const float* src = Bf + (size_t)(bn + r) * ldb + kk + h * 32;
        #pragma unroll
        for (int c = 0; c < 4; c++)
          *(v8bf*)&Bs[buf][r][h * 32 + c * 8] = cvt8_f32(src + c * 8);
      }
    } else {
      const int k  = tid >> 2;                     // 0..63
      const int n0 = (tid & 3) * 32;               // 32 consecutive n
      if constexpr (B_BF16) {
        const bf16_t* src = Bb + (size_t)(kk + k) * ldb + bn + n0;
        #pragma unroll
        for (int c = 0; c < 4; c++) {
          v8bf bvv = *(const v8bf*)(src + c * 8);
          #pragma unroll
          for (int j = 0; j < 8; j++) Bs[buf][n0 + c * 8 + j][k] = bvv[j];
        }
      } else {
        const float* src = Bf + (size_t)(kk + k) * ldb + bn + n0;
        #pragma unroll
        for (int j = 0; j < 32; j += 4) {
          float4 f = *(const float4*)(src + j);
          Bs[buf][n0 + j + 0][k] = (bf16_t)f.x;
          Bs[buf][n0 + j + 1][k] = (bf16_t)f.y;
          Bs[buf][n0 + j + 2][k] = (bf16_t)f.z;
          Bs[buf][n0 + j + 3][k] = (bf16_t)f.w;
        }
      }
    }
  };

  // ---- prologue: fill buffer 0 ----
  stageA(0, 0);
  stageB(0, 0);
  wait_async0();
  __syncthreads();

  const int nIter = K >> 6;
  for (int it = 0; it < nIter; ++it) {
    const int buf = it & 1;
    // prefetch next tile into the other buffer (DMA overlaps the WMMAs below)
    if (it + 1 < nIter) {
      stageA(buf ^ 1, (it + 1) << 6);
      stageB(buf ^ 1, (it + 1) << 6);
    }

    // ---- compute: 2 K-substeps of 32 -> 16 WMMAs ----
    #pragma unroll
    for (int ks = 0; ks < 64; ks += 32) {
      v16bf afrag[2], bfrag[4];
      #pragma unroll
      for (int mi = 0; mi < 2; mi++) {
        const int row = waveM * 32 + mi * 16 + lrow;
        v8bf lo = *(const v8bf*)&As[buf][row][ks + k0];        // K = ks+k0..+7
        v8bf hi = *(const v8bf*)&As[buf][row][ks + k0 + 16];   // K = ks+k0+16..
        #pragma unroll
        for (int e = 0; e < 8; e++) { afrag[mi][e] = lo[e]; afrag[mi][e + 8] = hi[e]; }
      }
      #pragma unroll
      for (int ni = 0; ni < 4; ni++) {
        const int col = waveN * 64 + ni * 16 + lrow;
        v8bf lo = *(const v8bf*)&Bs[buf][col][ks + half * 8];
        v8bf hi = *(const v8bf*)&Bs[buf][col][ks + half * 8 + 16];
        #pragma unroll
        for (int e = 0; e < 8; e++) { bfrag[ni][e] = lo[e]; bfrag[ni][e + 8] = hi[e]; }
      }
      #pragma unroll
      for (int mi = 0; mi < 2; mi++)
        #pragma unroll
        for (int ni = 0; ni < 4; ni++)
          acc[mi][ni] = __builtin_amdgcn_wmma_f32_16x16x32_bf16(
              false, afrag[mi], false, bfrag[ni], (short)0, acc[mi][ni],
              false, false);
    }

    // async DMA for next buffer must land; own frag reads already consumed
    wait_async0();
    __syncthreads();
  }

  // ---- fused epilogue: scale, bias, residual, relu, dual-precision store ----
  #pragma unroll
  for (int mi = 0; mi < 2; mi++) {
    #pragma unroll
    for (int ni = 0; ni < 4; ni++) {
      const int coln = bn + waveN * 64 + ni * 16 + lrow;
      const float bv = bias ? bias[coln] : 0.f;
      #pragma unroll
      for (int r = 0; r < 8; r++) {
        const int row = bm + waveM * 32 + mi * 16 + half * 8 + r;
        float v = acc[mi][ni][r] * scale + bv;
        if (resid) v += resid[(size_t)row * N + coln];
        if constexpr (RELU) v = fmaxf(v, 0.f);
        if (outF) outF[(size_t)row * N + coln] = v;
        if (outB) outB[(size_t)row * N + coln] = (bf16_t)v;
      }
    }
  }
}

// ---------------------------- causal softmax --------------------------------
// One block per row; fp32 attn map in place (d_out) + bf16 copy for attn @ v.
__global__ __launch_bounds__(256) void softmax_causal_kernel(
    float* __restrict__ sc, bf16_t* __restrict__ attnB)
{
  __shared__ float red[256];
  const int row = blockIdx.x;
  const int t = threadIdx.x;
  float* rp = sc + (size_t)row * S_LEN;

  float vals[S_LEN / 256];
  float m = -3.402823466e38f;
  #pragma unroll
  for (int i = 0; i < S_LEN / 256; i++) {
    const int c = t + i * 256;
    float v = (c <= row) ? rp[c] : -3.402823466e38f;
    vals[i] = v;
    m = fmaxf(m, v);
  }
  red[t] = m; __syncthreads();
  #pragma unroll
  for (int s2 = 128; s2 > 0; s2 >>= 1) {
    if (t < s2) red[t] = fmaxf(red[t], red[t + s2]);
    __syncthreads();
  }
  m = red[0]; __syncthreads();

  float sum = 0.f;
  #pragma unroll
  for (int i = 0; i < S_LEN / 256; i++) {
    const int c = t + i * 256;
    float e = (c <= row) ? __expf(vals[i] - m) : 0.f;
    vals[i] = e;
    sum += e;
  }
  red[t] = sum; __syncthreads();
  #pragma unroll
  for (int s2 = 128; s2 > 0; s2 >>= 1) {
    if (t < s2) red[t] += red[t + s2];
    __syncthreads();
  }
  const float inv = 1.f / red[0];
  #pragma unroll
  for (int i = 0; i < S_LEN / 256; i++) {
    const int c = t + i * 256;
    float p = vals[i] * inv;
    rp[c] = p;
    attnB[(size_t)row * S_LEN + c] = (bf16_t)p;
  }
}

// ---------------------------- log softmax -----------------------------------
__global__ __launch_bounds__(256) void log_softmax_kernel(
    const float* __restrict__ logits, float* __restrict__ out)
{
  __shared__ float red[256];
  const int row = blockIdx.x;
  const int t = threadIdx.x;
  const float* rp = logits + (size_t)row * C_DIM;

  float vals[C_DIM / 256];
  float m = -3.402823466e38f;
  #pragma unroll
  for (int i = 0; i < C_DIM / 256; i++) {
    vals[i] = rp[t + i * 256];
    m = fmaxf(m, vals[i]);
  }
  red[t] = m; __syncthreads();
  #pragma unroll
  for (int s2 = 128; s2 > 0; s2 >>= 1) {
    if (t < s2) red[t] = fmaxf(red[t], red[t + s2]);
    __syncthreads();
  }
  m = red[0]; __syncthreads();

  float sum = 0.f;
  #pragma unroll
  for (int i = 0; i < C_DIM / 256; i++) sum += __expf(vals[i] - m);
  red[t] = sum; __syncthreads();
  #pragma unroll
  for (int s2 = 128; s2 > 0; s2 >>= 1) {
    if (t < s2) red[t] += red[t + s2];
    __syncthreads();
  }
  const float lse = m + logf(red[0]);
  #pragma unroll
  for (int i = 0; i < C_DIM / 256; i++)
    out[(size_t)row * C_DIM + t + i * 256] = vals[i] - lse;
}

// ---------------------------- driver ----------------------------------------
extern "C" void kernel_launch(void* const* d_in, const int* in_sizes, int n_in,
                              void* d_out, int out_size, void* d_ws, size_t ws_size,
                              hipStream_t stream)
{
  (void)in_sizes; (void)n_in; (void)out_size; (void)ws_size;

  const int*   idx  = (const int*)  d_in[0];
  const float* emb  = (const float*)d_in[1];
  const float* pos  = (const float*)d_in[2];
  const float* Wq   = (const float*)d_in[3];
  const float* bq   = (const float*)d_in[4];
  const float* Wk   = (const float*)d_in[5];
  const float* bk   = (const float*)d_in[6];
  const float* Wv   = (const float*)d_in[7];
  const float* bv   = (const float*)d_in[8];
  const float* Wo   = (const float*)d_in[9];
  const float* bo   = (const float*)d_in[10];
  const float* W1   = (const float*)d_in[11];
  const float* b1   = (const float*)d_in[12];
  const float* W2   = (const float*)d_in[13];
  const float* b2   = (const float*)d_in[14];
  const float* Wout = (const float*)d_in[15];
  const float* bout = (const float*)d_in[16];

  float* out_logp = (float*)d_out;                        // [S, C]
  float* out_attn = out_logp + (size_t)S_LEN * C_DIM;     // [L, S, S]

  // workspace carve-out (~200 MB total)
  char* base = (char*)d_ws;
  size_t off = 0;
  auto carve = [&](size_t bytes) -> void* {
    void* p = base + off;
    off = (off + bytes + 255) & ~((size_t)255);
    return p;
  };
  float*  xF     = (float*) carve((size_t)S_LEN * D_DIM * 4);
  float*  rF     = (float*) carve((size_t)S_LEN * D_DIM * 4);
  bf16_t* xB     = (bf16_t*)carve((size_t)S_LEN * D_DIM * 2);
  bf16_t* rB     = (bf16_t*)carve((size_t)S_LEN * D_DIM * 2);
  bf16_t* qB     = (bf16_t*)carve((size_t)S_LEN * I_DIM * 2);  // reused: attn@v out
  bf16_t* kB     = (bf16_t*)carve((size_t)S_LEN * I_DIM * 2);  // reused: FFN hidden
  bf16_t* vB     = (bf16_t*)carve((size_t)S_LEN * I_DIM * 2);
  bf16_t* attnB  = (bf16_t*)carve((size_t)S_LEN * S_LEN * 2);
  float*  logits = (float*) carve((size_t)S_LEN * C_DIM * 4);

  const dim3 blk(256);
  const dim3 gI(I_DIM / 128, S_LEN / 128);
  const dim3 gS(S_LEN / 128, S_LEN / 128);
  const dim3 gD(D_DIM / 128, S_LEN / 128);
  const dim3 gC(C_DIM / 128, S_LEN / 128);
  const float inv_sqrt_I = 1.0f / 64.0f;   // 1/sqrt(4096)

  embed_kernel<<<S_LEN, blk, 0, stream>>>(idx, emb, pos, xF, xB);

  for (int l = 0; l < L_NUM; l++) {
    const float* Wq_l = Wq + (size_t)l * D_DIM * I_DIM;
    const float* Wk_l = Wk + (size_t)l * D_DIM * I_DIM;
    const float* Wv_l = Wv + (size_t)l * D_DIM * I_DIM;
    const float* W1_l = W1 + (size_t)l * D_DIM * I_DIM;
    const float* Wo_l = Wo + (size_t)l * I_DIM * D_DIM;
    const float* W2_l = W2 + (size_t)l * I_DIM * D_DIM;
    const float* bq_l = bq + (size_t)l * I_DIM;
    const float* bk_l = bk + (size_t)l * I_DIM;
    const float* bv_l = bv + (size_t)l * I_DIM;
    const float* b1_l = b1 + (size_t)l * I_DIM;
    const float* bo_l = bo + (size_t)l * D_DIM;
    const float* b2_l = b2 + (size_t)l * D_DIM;
    float* attn_l = out_attn + (size_t)l * S_LEN * S_LEN;

    // q/k/v = x @ W + b   (bf16 outputs)
    gemm_bf16_wmma<false, false, false><<<gI, blk, 0, stream>>>(
        xB, Wq_l, bq_l, nullptr, 1.f, nullptr, qB, I_DIM, D_DIM, D_DIM, I_DIM);
    gemm_bf16_wmma<false, false, false><<<gI, blk, 0, stream>>>(
        xB, Wk_l, bk_l, nullptr, 1.f, nullptr, kB, I_DIM, D_DIM, D_DIM, I_DIM);
    gemm_bf16_wmma<false, false, false><<<gI, blk, 0, stream>>>(
        xB, Wv_l, bv_l, nullptr, 1.f, nullptr, vB, I_DIM, D_DIM, D_DIM, I_DIM);

    // scores = (q @ k^T) / sqrt(I) -> fp32 straight into d_out attention map
    gemm_bf16_wmma<true, true, false><<<gS, blk, 0, stream>>>(
        qB, kB, nullptr, nullptr, inv_sqrt_I, attn_l, nullptr,
        S_LEN, I_DIM, I_DIM, I_DIM);

    // causal softmax in place (fp32) + bf16 copy for attn @ v
    softmax_causal_kernel<<<S_LEN, blk, 0, stream>>>(attn_l, attnB);

    // av = attn @ v  (bf16 out, reuses qB)
    gemm_bf16_wmma<true, false, false><<<gI, blk, 0, stream>>>(
        attnB, vB, nullptr, nullptr, 1.f, nullptr, qB,
        I_DIM, S_LEN, S_LEN, I_DIM);

    // r = x + av @ Wo + bo   (fp32 + bf16)
    gemm_bf16_wmma<false, false, false><<<gD, blk, 0, stream>>>(
        qB, Wo_l, bo_l, xF, 1.f, rF, rB, D_DIM, I_DIM, I_DIM, D_DIM);

    // h = relu(r @ W1 + b1)  (bf16 out, reuses kB)
    gemm_bf16_wmma<false, false, true><<<gI, blk, 0, stream>>>(
        rB, W1_l, b1_l, nullptr, 1.f, nullptr, kB, I_DIM, D_DIM, D_DIM, I_DIM);

    // x = r + h @ W2 + b2    (fp32 + bf16, becomes next layer input)
    gemm_bf16_wmma<false, false, false><<<gD, blk, 0, stream>>>(
        kB, W2_l, b2_l, rF, 1.f, xF, xB, D_DIM, I_DIM, I_DIM, D_DIM);
  }

  // logits = x @ Wout + bout ; then log_softmax -> d_out
  gemm_bf16_wmma<false, false, false><<<gC, blk, 0, stream>>>(
      xB, Wout, bout, nullptr, 1.f, logits, nullptr, C_DIM, D_DIM, D_DIM, C_DIM);
  log_softmax_kernel<<<S_LEN, blk, 0, stream>>>(logits, out_logp);
}
